// RadarTimeToFreqNet_67310727463470
// MI455X (gfx1250) — matrix operand order repaired and checked by reading, not compiled, not run on hardware
//
#include <hip/hip_runtime.h>
#include <math.h>

// ---------------------------------------------------------------------------
// RadarTimeToFreqNet for MI455X (gfx1250, wave32, WMMA).
// DFT stages run as complex GEMMs on v_wmma_f32_16x16x32_f16, with each wave
// producing a 16x64 output strip so A-fragments are reused 4x per K-step.
// ---------------------------------------------------------------------------

typedef __attribute__((ext_vector_type(16))) _Float16 v16h;
typedef __attribute__((ext_vector_type(8)))  _Float16 v8h;
typedef __attribute__((ext_vector_type(8)))  float    v8f;

#define EPSBN 1e-5f

// problem dims
#define NB 16
#define NRX 4
#define NC 128   // chirps (K of doppler DFT, also D=128)
#define NS 256   // samples (K of range DFT, also R=256)
#define ND 128
#define NR 256

// ---------------- workspace layout (bytes) ----------------
static constexpr size_t OFF_STATS = 0;                       // 4096 B
static constexpr size_t SZ_AX  = 2048ull * 256 * 2;          // f16 A matrices
static constexpr size_t OFF_AXR = 4096;
static constexpr size_t OFF_AXI = OFF_AXR + SZ_AX;
static constexpr size_t SZ_BT  = 256ull * 256 * 2;           // f16 range twiddles (transposed)
static constexpr size_t OFF_BTR = OFF_AXI + SZ_AX;
static constexpr size_t OFF_BTI = OFF_BTR + SZ_BT;
static constexpr size_t SZ_AD  = 128ull * 128 * 2;           // f16 doppler twiddles (transposed)
static constexpr size_t OFF_ADR = OFF_BTI + SZ_BT;
static constexpr size_t OFF_ADI = OFF_ADR + SZ_AD;
static constexpr size_t SZ_RT  = 16ull * 256 * 128 * 2;      // f16 range output, [b][r][c]
static constexpr size_t OFF_RRT = OFF_ADI + SZ_AD;
static constexpr size_t OFF_RIT = OFF_RRT + SZ_RT;
static constexpr size_t SZ_RD  = 16ull * 2 * 128 * 256 * 4;  // f32 range-doppler map
static constexpr size_t OFF_RD  = OFF_RIT + SZ_RT;
// big aliased regions
static constexpr size_t SZ_A   = 16ull * 32 * 128 * 256 * 4; // 67MB: conv1 out / e1 / u1
static constexpr size_t OFF_A  = OFF_RD + SZ_RD;
static constexpr size_t SZ_B   = 16ull * 128 * 32 * 64 * 4;  // 16.8MB: conv2 out / e3
static constexpr size_t OFF_B  = OFF_A + SZ_A;
static constexpr size_t SZ_P1  = 16ull * 32 * 64 * 128 * 4;  // 16.8MB: p1 / d2
static constexpr size_t OFF_P1 = OFF_B + SZ_B;
static constexpr size_t SZ_E2  = 16ull * 64 * 64 * 128 * 4;  // 33.6MB: e2 / u2 (in-place)
static constexpr size_t OFF_E2 = OFF_P1 + SZ_P1;
static constexpr size_t SZ_P2  = 16ull * 64 * 32 * 64 * 4;   // 8.4MB: p2 / d3
static constexpr size_t OFF_P2 = OFF_E2 + SZ_E2;
static constexpr size_t OFF_D1 = OFF_P2 + SZ_P2;             // 33.6MB: d1
// total ~185 MB

// stats float offsets
#define ST_TP1 0
#define ST_TP2 16
#define ST_E1  32
#define ST_E2  96
#define ST_E3  224
#define ST_D3  480
#define ST_D2  608
#define ST_D1  672

// ===========================================================================
// Temporal conv stage
// ===========================================================================
__global__ void tconv1_kernel(const float* __restrict__ x, const float* __restrict__ w1,
                              const float* __restrict__ b1, float* __restrict__ out1) {
  long idx = (long)blockIdx.x * blockDim.x + threadIdx.x;  // 2,097,152
  int s = (int)(idx & (NS - 1));
  long t = idx >> 8;
  int c = (int)(t & (NC - 1)); t >>= 7;
  int rx = (int)(t & 3);
  int b = (int)(t >> 2);
  const float* row = x + ((((long)b * NRX + rx) * NC + c) * NS) * 2;
  float xr[3], xi[3];
#pragma unroll
  for (int k = 0; k < 3; ++k) {
    int ss = s + k - 1;
    bool ok = (ss >= 0) && (ss < NS);
    xr[k] = ok ? row[ss * 2 + 0] : 0.f;
    xi[k] = ok ? row[ss * 2 + 1] : 0.f;
  }
#pragma unroll
  for (int o = 0; o < 8; ++o) {
    float acc = b1[o];
#pragma unroll
    for (int k = 0; k < 3; ++k)
      acc += xr[k] * w1[(o * 2 + 0) * 3 + k] + xi[k] * w1[(o * 2 + 1) * 3 + k];
    out1[((((long)b * 8 + o) * NRX + rx) * NC + c) * NS + s] = acc;
  }
}

// Deterministic per-channel BN statistics: one block per channel, tree reduce.
__global__ void channel_stats_kernel(const float* __restrict__ xbuf, float* __restrict__ stats,
                                     int C, long inner, int B) {
  __shared__ float ssum[256], ssq[256];
  const int c = blockIdx.x, tid = threadIdx.x;
  float s = 0.f, q = 0.f;
  for (int b = 0; b < B; ++b) {
    const float* p = xbuf + ((long)b * C + c) * inner;
    for (long i = tid; i < inner; i += 256) { float v = p[i]; s += v; q += v * v; }
  }
  ssum[tid] = s; ssq[tid] = q;
  __syncthreads();
  for (int off = 128; off > 0; off >>= 1) {
    if (tid < off) { ssum[tid] += ssum[tid + off]; ssq[tid] += ssq[tid + off]; }
    __syncthreads();
  }
  if (tid == 0) { stats[2 * c] = ssum[0]; stats[2 * c + 1] = ssq[0]; }
}

__global__ void tconv2_kernel(const float* __restrict__ out1, const float* __restrict__ stats,
                              const float* __restrict__ g1, const float* __restrict__ be1,
                              const float* __restrict__ w2, const float* __restrict__ b2,
                              float* __restrict__ out2) {
  const float invN = 1.0f / 2097152.0f;
  long idx = (long)blockIdx.x * blockDim.x + threadIdx.x;
  int s = (int)(idx & (NS - 1));
  long t = idx >> 8;
  int c = (int)(t & (NC - 1)); t >>= 7;
  int rx = (int)(t & 3);
  int b = (int)(t >> 2);
  float sc[8], sh[8];
#pragma unroll
  for (int i = 0; i < 8; ++i) {
    float mean = stats[2 * i] * invN;
    float var = stats[2 * i + 1] * invN - mean * mean;
    float inv = rsqrtf(var + EPSBN);
    sc[i] = g1[i] * inv;
    sh[i] = be1[i] - mean * sc[i];
  }
  float acc0 = b2[0], acc1 = b2[1];
#pragma unroll
  for (int i = 0; i < 8; ++i) {
    const float* row = out1 + (((long)b * 8 + i) * NRX + rx) * (long)NC * NS + (long)c * NS;
#pragma unroll
    for (int k = 0; k < 3; ++k) {
      int ss = s + k - 1;
      float v = 0.f;
      if (ss >= 0 && ss < NS) {
        v = row[ss] * sc[i] + sh[i];
        v = v > 0.f ? v : 0.f;
      }
      acc0 += v * w2[(0 * 8 + i) * 3 + k];
      acc1 += v * w2[(1 * 8 + i) * 3 + k];
    }
  }
  out2[((((long)b * 2 + 0) * NRX + rx) * NC + c) * NS + s] = acc0;
  out2[((((long)b * 2 + 1) * NRX + rx) * NC + c) * NS + s] = acc1;
}

__global__ void rxsum_kernel(const float* __restrict__ out2, const float* __restrict__ stats,
                             const float* __restrict__ g2, const float* __restrict__ be2,
                             _Float16* __restrict__ Axr, _Float16* __restrict__ Axi) {
  const float invN = 1.0f / 2097152.0f;
  long idx = (long)blockIdx.x * blockDim.x + threadIdx.x;  // 524,288
  int s = (int)(idx & (NS - 1));
  long t = idx >> 8;
  int c = (int)(t & (NC - 1));
  int b = (int)(t >> 7);
  float res[2];
#pragma unroll
  for (int ch = 0; ch < 2; ++ch) {
    float mean = stats[2 * ch] * invN;
    float var = stats[2 * ch + 1] * invN - mean * mean;
    float inv = rsqrtf(var + EPSBN);
    float sc = g2[ch] * inv, sh = be2[ch] - mean * sc;
    float acc = 0.f;
#pragma unroll
    for (int rx = 0; rx < NRX; ++rx) {
      float v = out2[((((long)b * 2 + ch) * NRX + rx) * NC + c) * NS + s] * sc + sh;
      acc += v > 0.f ? v : 0.f;
    }
    res[ch] = acc;
  }
  long m = (long)b * NC + c;
  Axr[m * NS + s] = (_Float16)res[0];
  Axi[m * NS + s] = (_Float16)res[1];
}

__global__ void twiddle_prep_kernel(const float* __restrict__ range_w, const float* __restrict__ doppler_w,
                                    _Float16* __restrict__ Btr, _Float16* __restrict__ Bti,
                                    _Float16* __restrict__ Adr, _Float16* __restrict__ Adi) {
  long idx = (long)blockIdx.x * blockDim.x + threadIdx.x;  // 65536
  int k = (int)(idx >> 8);       // S
  int n = (int)(idx & 255);      // R
  Btr[(long)n * NS + k] = (_Float16)range_w[((long)k * NR + n) * 2 + 0];
  Bti[(long)n * NS + k] = (_Float16)range_w[((long)k * NR + n) * 2 + 1];
  if (idx < (long)NC * ND) {
    int c = (int)(idx >> 7);     // C
    int d = (int)(idx & 127);    // D
    Adr[(long)d * NC + c] = (_Float16)doppler_w[((long)c * ND + d) * 2 + 0];
    Adi[(long)d * NC + c] = (_Float16)doppler_w[((long)c * ND + d) * 2 + 1];
  }
}

// ===========================================================================
// WMMA complex GEMMs (16x64 strip per wave: A fragments reused over 4 N-tiles)
// ===========================================================================
__device__ __forceinline__ v8f wmma16(v16h a, v16h b, v8f c) {
  return __builtin_amdgcn_wmma_f32_16x16x32_f16(false, a, false, b, (short)0, c, false, false);
}
// 16-bit A 16x32 fragment: lane holds row (lane&15), K chunk
// {kb..kb+15} with kb = (lane>>4)*16. Two 128-bit loads.
__device__ __forceinline__ v16h ldA(const _Float16* p) {
  v8h lo = *(const v8h*)p;
  v8h hi = *(const v8h*)(p + 8);
  return __builtin_shufflevector(lo, hi, 0, 1, 2, 3, 4, 5, 6, 7, 8, 9, 10, 11, 12, 13, 14, 15);
}

// rr/ri = (xr + j*xi) * (Wr + j*Wi): M=2048, N=256, K=256. Output stored
// transposed [b][r][c] f16 so the doppler GEMM's B-fragments are contiguous.
__global__ void range_gemm_kernel(const _Float16* __restrict__ Axr, const _Float16* __restrict__ Axi,
                                  const _Float16* __restrict__ Btr, const _Float16* __restrict__ Bti,
                                  _Float16* __restrict__ rrT, _Float16* __restrict__ riT) {
  const int tid = threadIdx.x, lane = tid & 31, wave = tid >> 5;
  const int tile = blockIdx.x * 8 + wave;   // 512 wave-strips (128 mt x 4 ng)
  const int mt = tile >> 2, ng = tile & 3;
  const int mbase = mt << 4, nbase0 = ng << 6;
  const int arow = mbase + (lane & 15);
  const int akb = (lane >> 4) << 4;
  const int bcol0 = nbase0 + (lane & 15);
  const int bkb = (lane >> 4) << 4;
  const _Float16* pAr = Axr + (long)arow * NS + akb;
  const _Float16* pAi = Axi + (long)arow * NS + akb;
  const _Float16* pBr = Btr + (long)bcol0 * NS + bkb;
  const _Float16* pBi = Bti + (long)bcol0 * NS + bkb;
  v8f cr[4] = {}, ci[4] = {};
  for (int k0 = 0; k0 < NS; k0 += 32) {
    __builtin_prefetch(pAr + k0 + 32, 0, 1);
    __builtin_prefetch(pAi + k0 + 32, 0, 1);
    v16h ar = ldA(pAr + k0);
    v16h ai = ldA(pAi + k0);
    v16h ain = -ai;
#pragma unroll
    for (int j = 0; j < 4; ++j) {
      const long joff = (long)j * 16 * NS;
      v16h br = *(const v16h*)(pBr + joff + k0);
      v16h bi = *(const v16h*)(pBi + joff + k0);
      cr[j] = wmma16(ar, br, cr[j]);
      cr[j] = wmma16(ain, bi, cr[j]);
      ci[j] = wmma16(ar, bi, ci[j]);
      ci[j] = wmma16(ai, br, ci[j]);
    }
  }
  const int nl = lane & 15, hiRow = (lane >> 4) * 8;
#pragma unroll
  for (int j = 0; j < 4; ++j) {
    const int nbase = nbase0 + j * 16;
#pragma unroll
    for (int g2 = 0; g2 < 8; ++g2) {
      int m = mbase + g2 + hiRow;
      int bb = m >> 7, cch = m & 127;
      long o = ((long)bb * NR + nbase + nl) * NC + cch;
      rrT[o] = (_Float16)cr[j][g2];
      riT[o] = (_Float16)ci[j][g2];
    }
  }
}

// rd = (Dr + j*Di)^T * (rr + j*ri) per batch: M=128(d), N=256(r), K=128(c).
__global__ void doppler_gemm_kernel(const _Float16* __restrict__ Adr, const _Float16* __restrict__ Adi,
                                    const _Float16* __restrict__ BrT, const _Float16* __restrict__ BiT,
                                    float* __restrict__ rd) {
  const int tid = threadIdx.x, lane = tid & 31, wave = tid >> 5;
  const int b = blockIdx.y;
  const int tile = blockIdx.x * 8 + wave;   // 32 wave-strips (8 mt x 4 ng)
  const int mt = tile >> 2, ng = tile & 3;
  const int mbase = mt << 4, nbase0 = ng << 6;
  const int arow = mbase + (lane & 15);
  const int akb = (lane >> 4) << 4;
  const int bcol0 = nbase0 + (lane & 15);
  const int bkb = (lane >> 4) << 4;
  const _Float16* pAr = Adr + (long)arow * NC + akb;
  const _Float16* pAi = Adi + (long)arow * NC + akb;
  const _Float16* pBr = BrT + ((long)b * NR + bcol0) * NC + bkb;
  const _Float16* pBi = BiT + ((long)b * NR + bcol0) * NC + bkb;
  v8f cr[4] = {}, ci[4] = {};
  for (int k0 = 0; k0 < NC; k0 += 32) {
    v16h ar = ldA(pAr + k0);
    v16h ai = ldA(pAi + k0);
    v16h ain = -ai;
#pragma unroll
    for (int j = 0; j < 4; ++j) {
      const long joff = (long)j * 16 * NC;
      v16h br = *(const v16h*)(pBr + joff + k0);
      v16h bi = *(const v16h*)(pBi + joff + k0);
      cr[j] = wmma16(ar, br, cr[j]);
      cr[j] = wmma16(ain, bi, cr[j]);
      ci[j] = wmma16(ar, bi, ci[j]);
      ci[j] = wmma16(ai, br, ci[j]);
    }
  }
  const int nl = lane & 15, hiRow = (lane >> 4) * 8;
#pragma unroll
  for (int j = 0; j < 4; ++j) {
    const int nbase = nbase0 + j * 16;
#pragma unroll
    for (int g2 = 0; g2 < 8; ++g2) {
      int m = mbase + g2 + hiRow;
      int n = nbase + nl;
      rd[(((long)b * 2 + 0) * ND + m) * NR + n] = cr[j][g2];
      rd[(((long)b * 2 + 1) * ND + m) * NR + n] = ci[j][g2];
    }
  }
}

// ===========================================================================
// U-Net kernels
// ===========================================================================
#define TW 32
#define TH 8
#define CKC 16

__global__ void conv3x3_kernel(const float* __restrict__ in, const float* __restrict__ w,
                               const float* __restrict__ bias, float* __restrict__ out,
                               int Cin, int Cout, int H, int W) {
  __shared__ float smem[CKC * (TH + 2) * (TW + 2)];
  __shared__ float wsh[8 * CKC * 9];
  const int tid = threadIdx.x;
  const int tx = tid & (TW - 1);
  const int ty = tid >> 5;
  const int x0 = blockIdx.x * TW;
  const int y0 = blockIdx.y * TH;
  const int ngr = Cout >> 3;
  const int og = blockIdx.z % ngr;
  const int b = blockIdx.z / ngr;
  const int cout0 = og * 8;
  const int x = x0 + tx, y = y0 + ty;

  float acc[8];
#pragma unroll
  for (int o = 0; o < 8; ++o) acc[o] = bias[cout0 + o];

  for (int cin0 = 0; cin0 < Cin; cin0 += CKC) {
    const int cc = (Cin - cin0 < CKC) ? (Cin - cin0) : CKC;
    const int tileN = cc * (TH + 2) * (TW + 2);
    for (int i = tid; i < tileN; i += 256) {
      int ci = i / ((TH + 2) * (TW + 2));
      int rem = i - ci * ((TH + 2) * (TW + 2));
      int yy = rem / (TW + 2), xx = rem - yy * (TW + 2);
      int gy = y0 - 1 + yy, gx = x0 - 1 + xx;
      float v = 0.f;
      if (gy >= 0 && gy < H && gx >= 0 && gx < W)
        v = in[(((long)b * Cin + cin0 + ci) * H + gy) * (long)W + gx];
      smem[i] = v;
    }
    const int wN = 8 * cc * 9;
    for (int i = tid; i < wN; i += 256) {
      int o = i / (cc * 9);
      int rem = i - o * (cc * 9);
      int ci = rem / 9, kk = rem - ci * 9;
      wsh[(o * CKC + ci) * 9 + kk] = w[((long)(cout0 + o) * Cin + cin0 + ci) * 9 + kk];
    }
    __syncthreads();
    for (int ci = 0; ci < cc; ++ci) {
      const float* sp = &smem[ci * (TH + 2) * (TW + 2)];
#pragma unroll
      for (int ky = 0; ky < 3; ++ky)
#pragma unroll
        for (int kx = 0; kx < 3; ++kx) {
          float v = sp[(ty + ky) * (TW + 2) + tx + kx];
#pragma unroll
          for (int o = 0; o < 8; ++o)
            acc[o] += v * wsh[(o * CKC + ci) * 9 + ky * 3 + kx];
        }
    }
    __syncthreads();
  }
#pragma unroll
  for (int o = 0; o < 8; ++o)
    out[(((long)b * Cout + cout0 + o) * H + y) * (long)W + x] = acc[o];
}

__global__ void bn_relu_kernel(float* __restrict__ xbuf, const float* __restrict__ stats,
                               const float* __restrict__ g, const float* __restrict__ be,
                               int C, long HW, long total, float invN) {
  long idx = (long)blockIdx.x * blockDim.x + threadIdx.x;
  if (idx >= total) return;
  int c = (int)((idx / HW) % C);
  float mean = stats[2 * c] * invN;
  float var = stats[2 * c + 1] * invN - mean * mean;
  float inv = rsqrtf(var + EPSBN);
  float sc = g[c] * inv;
  float sh = be[c] - mean * sc;
  float v = xbuf[idx] * sc + sh;
  xbuf[idx] = v > 0.f ? v : 0.f;
}

__global__ void pool2_kernel(const float* __restrict__ in, float* __restrict__ out,
                             long planes, int H, int W) {
  int ow = W >> 1, oh = H >> 1;
  long total = planes * oh * ow;
  long idx = (long)blockIdx.x * blockDim.x + threadIdx.x;
  if (idx >= total) return;
  int xx = (int)(idx % ow);
  long t = idx / ow;
  int yy = (int)(t % oh);
  long p = t / oh;
  const float* ip = in + (p * H + 2 * yy) * (long)W + 2 * xx;
  out[idx] = fmaxf(fmaxf(ip[0], ip[1]), fmaxf(ip[W], ip[W + 1]));
}

__global__ void upsample_add_kernel(const float* __restrict__ d, float* __restrict__ e_inout,
                                    long planes, int h, int w) {
  int H2 = 2 * h, W2 = 2 * w;
  long total = planes * (long)H2 * W2;
  long idx = (long)blockIdx.x * blockDim.x + threadIdx.x;
  if (idx >= total) return;
  int x2 = (int)(idx % W2);
  long t = idx / W2;
  int y2 = (int)(t % H2);
  long p = t / H2;
  float sy = (float)(h - 1) / (float)(H2 - 1);
  float sx = (float)(w - 1) / (float)(W2 - 1);
  float py = y2 * sy, px = x2 * sx;
  int ly = (int)py, lx = (int)px;
  int hy = (ly + 1 < h) ? ly + 1 : h - 1;
  int hx = (lx + 1 < w) ? lx + 1 : w - 1;
  float wy = py - (float)ly, wx = px - (float)lx;
  const float* dp = d + p * (long)h * w;
  float v = dp[(long)ly * w + lx] * (1.f - wy) * (1.f - wx)
          + dp[(long)ly * w + hx] * (1.f - wy) * wx
          + dp[(long)hy * w + lx] * wy * (1.f - wx)
          + dp[(long)hy * w + hx] * wy * wx;
  e_inout[idx] += v;
}

__global__ void outconv_sigmoid_kernel(const float* __restrict__ d1, const float* __restrict__ ow,
                                       const float* __restrict__ ob, float* __restrict__ out) {
  long idx = (long)blockIdx.x * blockDim.x + threadIdx.x;  // 524,288 == B*D*R
  int xx = (int)(idx & 255);
  long t = idx >> 8;
  int yy = (int)(t & 127);
  int b = (int)(t >> 7);
  float acc = ob[0];
  const float* p = d1 + (((long)b * 16) * ND + yy) * (long)NR + xx;
#pragma unroll
  for (int ch = 0; ch < 16; ++ch) acc += p[(long)ch * ND * NR] * ow[ch];
  out[idx] = 1.f / (1.f + expf(-acc));
}

// ===========================================================================
extern "C" void kernel_launch(void* const* d_in, const int* in_sizes, int n_in,
                              void* d_out, int out_size, void* d_ws, size_t ws_size,
                              hipStream_t stream) {
  (void)in_sizes; (void)n_in; (void)out_size; (void)ws_size;
  const float* x       = (const float*)d_in[0];
  const float* tp_w1   = (const float*)d_in[1];
  const float* tp_b1   = (const float*)d_in[2];
  const float* tp_g1   = (const float*)d_in[3];
  const float* tp_be1  = (const float*)d_in[4];
  const float* tp_w2   = (const float*)d_in[5];
  const float* tp_b2   = (const float*)d_in[6];
  const float* tp_g2   = (const float*)d_in[7];
  const float* tp_be2  = (const float*)d_in[8];
  const float* range_w = (const float*)d_in[9];
  const float* dopp_w  = (const float*)d_in[10];
  const float* enc1_w = (const float*)d_in[11]; const float* enc1_b = (const float*)d_in[12];
  const float* enc1_g = (const float*)d_in[13]; const float* enc1_be = (const float*)d_in[14];
  const float* enc2_w = (const float*)d_in[15]; const float* enc2_b = (const float*)d_in[16];
  const float* enc2_g = (const float*)d_in[17]; const float* enc2_be = (const float*)d_in[18];
  const float* enc3_w = (const float*)d_in[19]; const float* enc3_b = (const float*)d_in[20];
  const float* enc3_g = (const float*)d_in[21]; const float* enc3_be = (const float*)d_in[22];
  const float* dec3_w = (const float*)d_in[23]; const float* dec3_b = (const float*)d_in[24];
  const float* dec3_g = (const float*)d_in[25]; const float* dec3_be = (const float*)d_in[26];
  const float* dec2_w = (const float*)d_in[27]; const float* dec2_b = (const float*)d_in[28];
  const float* dec2_g = (const float*)d_in[29]; const float* dec2_be = (const float*)d_in[30];
  const float* dec1_w = (const float*)d_in[31]; const float* dec1_b = (const float*)d_in[32];
  const float* dec1_g = (const float*)d_in[33]; const float* dec1_be = (const float*)d_in[34];
  const float* out_w  = (const float*)d_in[35]; const float* out_b  = (const float*)d_in[36];

  char* ws = (char*)d_ws;
  float*    stats = (float*)(ws + OFF_STATS);
  _Float16* Axr = (_Float16*)(ws + OFF_AXR);
  _Float16* Axi = (_Float16*)(ws + OFF_AXI);
  _Float16* Btr = (_Float16*)(ws + OFF_BTR);
  _Float16* Bti = (_Float16*)(ws + OFF_BTI);
  _Float16* Adr = (_Float16*)(ws + OFF_ADR);
  _Float16* Adi = (_Float16*)(ws + OFF_ADI);
  _Float16* rrT = (_Float16*)(ws + OFF_RRT);
  _Float16* riT = (_Float16*)(ws + OFF_RIT);
  float* rd    = (float*)(ws + OFF_RD);
  float* regA  = (float*)(ws + OFF_A);    // conv1 out -> e1 -> u1
  float* regB  = (float*)(ws + OFF_B);    // conv2 out -> e3
  float* regP1 = (float*)(ws + OFF_P1);   // p1 -> d2
  float* regE2 = (float*)(ws + OFF_E2);   // e2 -> u2 (in place)
  float* regP2 = (float*)(ws + OFF_P2);   // p2 -> d3
  float* regD1 = (float*)(ws + OFF_D1);   // d1

  // --- temporal stage ---
  tconv1_kernel<<<8192, 256, 0, stream>>>(x, tp_w1, tp_b1, regA);
  channel_stats_kernel<<<8, 256, 0, stream>>>(regA, stats + ST_TP1, 8, 4L * 128 * 256, 16);
  tconv2_kernel<<<8192, 256, 0, stream>>>(regA, stats + ST_TP1, tp_g1, tp_be1, tp_w2, tp_b2, regB);
  channel_stats_kernel<<<2, 256, 0, stream>>>(regB, stats + ST_TP2, 2, 4L * 128 * 256, 16);
  rxsum_kernel<<<2048, 256, 0, stream>>>(regB, stats + ST_TP2, tp_g2, tp_be2, Axr, Axi);

  // --- DFT stages (WMMA complex GEMMs) ---
  twiddle_prep_kernel<<<256, 256, 0, stream>>>(range_w, dopp_w, Btr, Bti, Adr, Adi);
  range_gemm_kernel<<<64, 256, 0, stream>>>(Axr, Axi, Btr, Bti, rrT, riT);
  doppler_gemm_kernel<<<dim3(4, 16), 256, 0, stream>>>(Adr, Adi, rrT, riT, rd);

  // --- U-Net ---
  conv3x3_kernel<<<dim3(8, 16, 64), 256, 0, stream>>>(rd, enc1_w, enc1_b, regA, 2, 32, 128, 256);
  channel_stats_kernel<<<32, 256, 0, stream>>>(regA, stats + ST_E1, 32, 32768, 16);
  bn_relu_kernel<<<65536, 256, 0, stream>>>(regA, stats + ST_E1, enc1_g, enc1_be, 32, 32768, 16L * 32 * 32768, 1.f / 524288.f);
  pool2_kernel<<<16384, 256, 0, stream>>>(regA, regP1, 16L * 32, 128, 256);
  conv3x3_kernel<<<dim3(4, 8, 128), 256, 0, stream>>>(regP1, enc2_w, enc2_b, regE2, 32, 64, 64, 128);
  channel_stats_kernel<<<64, 256, 0, stream>>>(regE2, stats + ST_E2, 64, 8192, 16);
  bn_relu_kernel<<<32768, 256, 0, stream>>>(regE2, stats + ST_E2, enc2_g, enc2_be, 64, 8192, 16L * 64 * 8192, 1.f / 131072.f);
  pool2_kernel<<<8192, 256, 0, stream>>>(regE2, regP2, 16L * 64, 64, 128);
  conv3x3_kernel<<<dim3(2, 4, 256), 256, 0, stream>>>(regP2, enc3_w, enc3_b, regB, 64, 128, 32, 64);
  channel_stats_kernel<<<128, 256, 0, stream>>>(regB, stats + ST_E3, 128, 2048, 16);
  bn_relu_kernel<<<16384, 256, 0, stream>>>(regB, stats + ST_E3, enc3_g, enc3_be, 128, 2048, 16L * 128 * 2048, 1.f / 32768.f);
  conv3x3_kernel<<<dim3(2, 4, 128), 256, 0, stream>>>(regB, dec3_w, dec3_b, regP2, 128, 64, 32, 64);
  channel_stats_kernel<<<64, 256, 0, stream>>>(regP2, stats + ST_D3, 64, 2048, 16);
  bn_relu_kernel<<<8192, 256, 0, stream>>>(regP2, stats + ST_D3, dec3_g, dec3_be, 64, 2048, 16L * 64 * 2048, 1.f / 32768.f);
  upsample_add_kernel<<<32768, 256, 0, stream>>>(regP2, regE2, 16L * 64, 32, 64);
  conv3x3_kernel<<<dim3(4, 8, 64), 256, 0, stream>>>(regE2, dec2_w, dec2_b, regP1, 64, 32, 64, 128);
  channel_stats_kernel<<<32, 256, 0, stream>>>(regP1, stats + ST_D2, 32, 8192, 16);
  bn_relu_kernel<<<16384, 256, 0, stream>>>(regP1, stats + ST_D2, dec2_g, dec2_be, 32, 8192, 16L * 32 * 8192, 1.f / 131072.f);
  upsample_add_kernel<<<65536, 256, 0, stream>>>(regP1, regA, 16L * 32, 64, 128);
  conv3x3_kernel<<<dim3(8, 16, 32), 256, 0, stream>>>(regA, dec1_w, dec1_b, regD1, 32, 16, 128, 256);
  channel_stats_kernel<<<16, 256, 0, stream>>>(regD1, stats + ST_D1, 16, 32768, 16);
  bn_relu_kernel<<<32768, 256, 0, stream>>>(regD1, stats + ST_D1, dec1_g, dec1_be, 16, 32768, 16L * 16 * 32768, 1.f / 524288.f);
  outconv_sigmoid_kernel<<<2048, 256, 0, stream>>>(regD1, out_w, out_b, (float*)d_out);
}
